// EmbedMatcher_19043884990788
// MI455X (gfx1250) — matrix operand order repaired and checked by reading, not compiled, not run on hardware
//
#include <hip/hip_runtime.h>
#include <hip/hip_bf16.h>

typedef __attribute__((ext_vector_type(16))) _Float16 v16h;
typedef __attribute__((ext_vector_type(8)))  _Float16 h8;
typedef __attribute__((ext_vector_type(8)))  float    v8f;

#define DDIM 128
#define MDIM 200
#define KSEL 32
#define BQ   1024
#define FEWN 5
#define DM   256
#define DF   512
#define HDIM 512
#define KMAX 512
#define LDPAD 8  // halfs; row stride K+8 -> 16B aligned, bank-conflict free

__device__ __forceinline__ float wsum(float v) {
#pragma unroll
  for (int m = 16; m > 0; m >>= 1) v += __shfl_xor(v, m, 32);
  return v;
}

// ---------------------------------------------------------------------------
// Neighbor encoder front half: cosine sims over M=200 neighbors, top-K=32
// (JAX top_k semantics: descending, lower index wins ties), mean of selected
// [rel | ent] concat -> cat (rows x 256).  One block (4 waves) per row.
// ---------------------------------------------------------------------------
__global__ __launch_bounds__(128) void neighbor_kernel(
    const int* __restrict__ conn, const int* __restrict__ ids, int idStride,
    int idOff, const float* __restrict__ emb, float* __restrict__ cat,
    int nrows) {
  int row = blockIdx.x;
  if (row >= nrows) return;
  int lane = threadIdx.x & 31;
  int wave = threadIdx.x >> 5;
  __shared__ float sim_s[224];
  if (threadIdx.x < 24) sim_s[200 + threadIdx.x] = -INFINITY;

  int cid = ids[row * idStride + idOff];
  float4 c4 = ((const float4*)(emb + (long)cid * DDIM))[lane];
  float cn2 = wsum(c4.x * c4.x + c4.y * c4.y + c4.z * c4.z + c4.w * c4.w);
  float cn = sqrtf(cn2);

  // sims: lanes cover D=128 (coalesced 512B row reads), waves cover m.
  for (int m = wave; m < MDIM; m += 4) {
    int eid = conn[((long)row * MDIM + m) * 2 + 1];
    float4 e4 = ((const float4*)(emb + (long)eid * DDIM))[lane];
    float num = wsum(c4.x * e4.x + c4.y * e4.y + c4.z * e4.z + c4.w * e4.w);
    float en2 = wsum(e4.x * e4.x + e4.y * e4.y + e4.z * e4.z + e4.w * e4.w);
    if (lane == 0) {
      float den = fmaxf(cn * sqrtf(en2), 1e-8f);
      sim_s[m] = num / den;
    }
  }
  __syncthreads();
  if (wave != 0) return;  // wave 0 does top-K + accumulation

  float vals[7];
#pragma unroll
  for (int j = 0; j < 7; ++j) vals[j] = sim_s[lane + 32 * j];
  unsigned taken = 0;
  float4 accR = {0.f, 0.f, 0.f, 0.f}, accE = {0.f, 0.f, 0.f, 0.f};
  for (int it = 0; it < KSEL; ++it) {
    float bv = -INFINITY;
    int bi = 0x7fffffff;
#pragma unroll
    for (int j = 0; j < 7; ++j) {
      if (!((taken >> j) & 1u)) {
        int idx = lane + 32 * j;
        float v = vals[j];
        if (v > bv || (v == bv && idx < bi)) { bv = v; bi = idx; }
      }
    }
#pragma unroll
    for (int m = 16; m > 0; m >>= 1) {
      float ov = __shfl_xor(bv, m, 32);
      int oi = __shfl_xor(bi, m, 32);
      if (ov > bv || (ov == bv && oi < bi)) { bv = ov; bi = oi; }
    }
    if ((bi & 31) == lane) taken |= 1u << (bi >> 5);
    int rid = conn[((long)row * MDIM + bi) * 2 + 0];
    int eid = conn[((long)row * MDIM + bi) * 2 + 1];
    float4 r4 = ((const float4*)(emb + (long)rid * DDIM))[lane];
    float4 e4 = ((const float4*)(emb + (long)eid * DDIM))[lane];
    accR.x += r4.x; accR.y += r4.y; accR.z += r4.z; accR.w += r4.w;
    accE.x += e4.x; accE.y += e4.y; accE.z += e4.z; accE.w += e4.w;
  }
  const float inv = 1.0f / (float)KSEL;
  float* o = cat + (long)row * DM;
  ((float4*)o)[lane] =
      make_float4(accR.x * inv, accR.y * inv, accR.z * inv, accR.w * inv);
  ((float4*)(o + DDIM))[lane] =
      make_float4(accE.x * inv, accE.y * inv, accE.z * inv, accE.w * inv);
}

// f32 -> f16 weight pre-conversion (halves B traffic, removes cvt from loop)
__global__ __launch_bounds__(256) void convert_f16_kernel(
    const float* __restrict__ src, _Float16* __restrict__ dst, long n) {
  long i = ((long)blockIdx.x * blockDim.x + threadIdx.x) * 4;
  if (i >= n) return;
  float4 v = *(const float4*)(src + i);
  dst[i + 0] = (_Float16)v.x;
  dst[i + 1] = (_Float16)v.y;
  dst[i + 2] = (_Float16)v.z;
  dst[i + 3] = (_Float16)v.w;
}

// ---------------------------------------------------------------------------
// WMMA GEMM:  C = act( A1@W1^T [+ A2@W2^T] + bias1 + bias2 )
// A* f32 (rows x K*), W* pre-converted f16 (nout x K*) row-major.
// Block = 4 waves, each owns a 16x16 tile (block: 16 rows x 64 cols).
// A strip staged in LDS ONCE per pair; inner loop is barrier-free
// ds_load_b128 + global b128 + v_wmma.  Requires nout % 64 == 0, K % 32 == 0.
// act: 0=none 1=tanh 2=relu.
// ---------------------------------------------------------------------------
__global__ __launch_bounds__(128) void gemm_wmma_kernel(
    const float* __restrict__ A1, int lda1, const _Float16* __restrict__ W1,
    int K1, const float* __restrict__ A2, int lda2,
    const _Float16* __restrict__ W2, int K2, float* __restrict__ C, int ldc,
    const float* __restrict__ bias1, const float* __restrict__ bias2, int rows,
    int nout, int act) {
  __shared__ _Float16 As[16 * (KMAX + LDPAD)];
  const int lane = threadIdx.x & 31;
  const int wave = threadIdx.x >> 5;
  const int rbase = blockIdx.x * 16;
  const int nbase = blockIdx.y * 64 + wave * 16;
  const int lrow = lane & 15;
  const int hi = lane >> 4;
  const int ncol = nbase + lrow;

  v8f acc;
#pragma unroll
  for (int r = 0; r < 8; ++r) acc[r] = 0.0f;

#pragma unroll 1
  for (int pair = 0; pair < 2; ++pair) {
    const float* A = pair ? A2 : A1;
    if (!A) break;
    const int lda = pair ? lda2 : lda1;
    const _Float16* W = pair ? W2 : W1;
    const int K = pair ? K2 : K1;
    const int ldAs = K + LDPAD;

    __syncthreads();  // protect As reuse across pairs
    {  // stage full 16 x K A strip as f16 (8 threads per row)
      int r = threadIdx.x >> 3;
      int t8 = threadIdx.x & 7;
      int gr = rbase + r;
      bool ok = gr < rows;
      for (int c = t8 * 4; c < K; c += 32) {
        float4 a4 = ok ? *(const float4*)(A + (long)gr * lda + c)
                       : make_float4(0.f, 0.f, 0.f, 0.f);
        As[r * ldAs + c + 0] = (_Float16)a4.x;
        As[r * ldAs + c + 1] = (_Float16)a4.y;
        As[r * ldAs + c + 2] = (_Float16)a4.z;
        As[r * ldAs + c + 3] = (_Float16)a4.w;
      }
    }
    __syncthreads();

    // A frag: lanes 0-15 (row, K kb+0..7 & kb+16..23); lanes 16-31 (+8).
    // B frag: W[ncol][kb + hi*8 + 0..7] & W[ncol][kb + 16 + hi*8 + 0..7].
    const _Float16* ap0 = As + lrow * ldAs + hi * 8;
    const _Float16* wp0 = W + (long)ncol * K + hi * 8;
#pragma unroll 2
    for (int kb = 0; kb < K; kb += 32) {
      h8 a_lo = *(const h8*)(ap0 + kb);
      h8 a_hi = *(const h8*)(ap0 + kb + 16);
      h8 b_lo = *(const h8*)(wp0 + kb);
      h8 b_hi = *(const h8*)(wp0 + kb + 16);
      v16h af, bf;
#pragma unroll
      for (int j = 0; j < 8; ++j) {
        af[j] = a_lo[j];
        af[j + 8] = a_hi[j];
        bf[j] = b_lo[j];
        bf[j + 8] = b_hi[j];
      }
      acc = __builtin_amdgcn_wmma_f32_16x16x32_f16(false, af, false, bf,
                                                   (short)0, acc, false, false);
    }
  }

  float bs = 0.0f;
  if (bias1) bs += bias1[ncol];
  if (bias2) bs += bias2[ncol];
#pragma unroll
  for (int r = 0; r < 8; ++r) {
    int rr = rbase + (hi ? 8 + r : r);
    if (rr < rows) {
      float v = acc[r] + bs;
      if (act == 1) v = tanhf(v);
      else if (act == 2) v = fmaxf(v, 0.0f);
      C[(long)rr * ldc + ncol] = v;
    }
  }
}

// ---------------------------------------------------------------------------
// Residual + LayerNorm over DM=256:  out = g*(y-mu)/sqrt(var+eps)+b, y=h+x
// ---------------------------------------------------------------------------
__global__ __launch_bounds__(32) void ln_kernel(
    const float* __restrict__ h, const float* __restrict__ x,
    const float* __restrict__ g, const float* __restrict__ b,
    float* __restrict__ out, int rows) {
  int row = blockIdx.x;
  if (row >= rows) return;
  int lane = threadIdx.x;
  float y[8];
  float s = 0.f;
#pragma unroll
  for (int j = 0; j < 8; ++j) {
    int c = lane + 32 * j;
    y[j] = h[(long)row * DM + c] + x[(long)row * DM + c];
    s += y[j];
  }
  float mu = wsum(s) * (1.0f / DM);
  float v = 0.f;
#pragma unroll
  for (int j = 0; j < 8; ++j) {
    float d = y[j] - mu;
    v += d * d;
  }
  v = wsum(v) * (1.0f / DM);
  float inv = rsqrtf(v + 1e-5f);
#pragma unroll
  for (int j = 0; j < 8; ++j) {
    int c = lane + 32 * j;
    out[(long)row * DM + c] = g[c] * (y[j] - mu) * inv + b[c];
  }
}

// support_g = mean over FEW rows of se_s; support_gn = l2norm(support_g)
__global__ __launch_bounds__(256) void support_mean_kernel(
    const float* __restrict__ ses, float* __restrict__ sg,
    float* __restrict__ sgn) {
  int c = threadIdx.x;
  float s = 0.f;
#pragma unroll
  for (int f = 0; f < FEWN; ++f) s += ses[f * DM + c];
  s *= (1.0f / FEWN);
  sg[c] = s;
  __shared__ float red[8];
  float ss = wsum(s * s);
  if ((c & 31) == 0) red[c >> 5] = ss;
  __syncthreads();
  float t = 0.f;
#pragma unroll
  for (int i = 0; i < 8; ++i) t += red[i];
  sgn[c] = s / fmaxf(sqrtf(t), 1e-12f);
}

// LSTM pointwise: c' = sig(f)*c + sig(i)*tanh(g); hf = sig(o)*tanh(c');
// h = qg + hf[:DM]; h_r = [h | support_g]  (attn over 1-row support == 1)
__global__ __launch_bounds__(256) void lstm_pw_kernel(
    const float* __restrict__ gates, float* __restrict__ cst,
    const float* __restrict__ qg, const float* __restrict__ sg,
    float* __restrict__ h, float* __restrict__ hr, int step, int rows) {
  long idx = (long)blockIdx.x * blockDim.x + threadIdx.x;
  long total = (long)rows * HDIM;
  if (idx >= total) return;
  int row = (int)(idx / HDIM), u = (int)(idx % HDIM);
  const float* gp = gates + (long)row * 4 * HDIM;
  float gi = gp[u], gf = gp[HDIM + u], gg = gp[2 * HDIM + u],
        go = gp[3 * HDIM + u];
  float cprev = step ? cst[idx] : 0.0f;
  float si = 1.f / (1.f + expf(-gi));
  float sf = 1.f / (1.f + expf(-gf));
  float so = 1.f / (1.f + expf(-go));
  float cn = sf * cprev + si * tanhf(gg);
  cst[idx] = cn;
  float hf = so * tanhf(cn);
  if (u < DM) {
    float hn = qg[(long)row * DM + u] + hf;
    h[(long)row * DM + u] = hn;
    hr[(long)row * HDIM + u] = hn;
  } else {
    hr[(long)row * HDIM + u] = sg[u - DM];
  }
}

// out[b] = dot(l2norm(h[b]), support_gn)
__global__ __launch_bounds__(32) void final_kernel(
    const float* __restrict__ h, const float* __restrict__ sgn,
    float* __restrict__ out, int rows) {
  int row = blockIdx.x;
  if (row >= rows) return;
  int lane = threadIdx.x;
  float d = 0.f, n2 = 0.f;
#pragma unroll
  for (int j = 0; j < 8; ++j) {
    int c = lane + 32 * j;
    float v = h[(long)row * DM + c];
    d += v * sgn[c];
    n2 += v * v;
  }
  d = wsum(d);
  n2 = wsum(n2);
  if (lane == 0) out[row] = d / fmaxf(sqrtf(n2), 1e-12f);
}

// ---------------------------------------------------------------------------
extern "C" void kernel_launch(void* const* d_in, const int* in_sizes, int n_in,
                              void* d_out, int out_size, void* d_ws,
                              size_t ws_size, hipStream_t stream) {
  const int* query = (const int*)d_in[0];
  const int* support = (const int*)d_in[1];
  const int* qlc = (const int*)d_in[2];
  const int* qrc = (const int*)d_in[4];
  const int* slc = (const int*)d_in[6];
  const int* srcn = (const int*)d_in[8];
  const float* emb = (const float*)d_in[10];
  const float* gcnW = (const float*)d_in[11];
  const float* gcnBias = (const float*)d_in[12];
  const float* gcnB = (const float*)d_in[13];
  const float* w1 = (const float*)d_in[14];
  const float* b1 = (const float*)d_in[15];
  const float* w2 = (const float*)d_in[16];
  const float* b2 = (const float*)d_in[17];
  const float* lng = (const float*)d_in[18];
  const float* lnb = (const float*)d_in[19];
  const float* Wih = (const float*)d_in[20];
  const float* Whh = (const float*)d_in[21];
  const float* bih = (const float*)d_in[22];
  const float* bhh = (const float*)d_in[23];

  char* ws = (char*)d_ws;
  size_t off = 0;
  auto alloc = [&](size_t bytes) -> void* {
    void* p = (void*)(ws + off);
    off += (bytes + 255) & ~(size_t)255;
    return p;
  };
  float* cat = (float*)alloc((size_t)BQ * DM * 4);
  float* QN = (float*)alloc((size_t)BQ * DM * 4);
  float* SN = (float*)alloc((size_t)FEWN * DM * 4);
  float* Hq = (float*)alloc((size_t)BQ * DF * 4);
  float* SEh = (float*)alloc((size_t)BQ * DM * 4);
  float* QG = (float*)alloc((size_t)BQ * DM * 4);
  float* Hs = (float*)alloc((size_t)FEWN * DF * 4);
  float* SEhs = (float*)alloc((size_t)FEWN * DM * 4);
  float* SES = (float*)alloc((size_t)FEWN * DM * 4);
  float* SG = (float*)alloc((size_t)DM * 4);
  float* SGN = (float*)alloc((size_t)DM * 4);
  float* gates = (float*)alloc((size_t)BQ * 4 * HDIM * 4);
  float* cst = (float*)alloc((size_t)BQ * HDIM * 4);
  float* hbuf = (float*)alloc((size_t)BQ * DM * 4);
  float* hr = (float*)alloc((size_t)BQ * HDIM * 4);
  _Float16* gcnW16 = (_Float16*)alloc((size_t)DDIM * DM * 2);
  _Float16* w116 = (_Float16*)alloc((size_t)DF * DM * 2);
  _Float16* w216 = (_Float16*)alloc((size_t)DM * DF * 2);
  _Float16* Wih16 = (_Float16*)alloc((size_t)4 * HDIM * DM * 2);
  _Float16* Whh16 = (_Float16*)alloc((size_t)4 * HDIM * HDIM * 2);
  (void)ws_size;

  auto conv = [&](const float* s, _Float16* d, long n) {
    convert_f16_kernel<<<(int)((n / 4 + 255) / 256), 256, 0, stream>>>(s, d, n);
  };
  conv(gcnW, gcnW16, (long)DDIM * DM);
  conv(w1, w116, (long)DF * DM);
  conv(w2, w216, (long)DM * DF);
  conv(Wih, Wih16, (long)4 * HDIM * DM);
  conv(Whh, Whh16, (long)4 * HDIM * HDIM);

  auto gemm = [&](const float* A1, int lda1, const _Float16* W1h, int K1,
                  const float* A2, int lda2, const _Float16* W2h, int K2,
                  float* C, int ldc, const float* bs1, const float* bs2,
                  int rows, int nout, int act) {
    dim3 grid((rows + 15) / 16, nout / 64);
    gemm_wmma_kernel<<<grid, 128, 0, stream>>>(A1, lda1, W1h, K1, A2, lda2,
                                               W2h, K2, C, ldc, bs1, bs2, rows,
                                               nout, act);
  };

  // ---- neighbor encoders (GCN matmul folded after mean-over-K) ----
  neighbor_kernel<<<BQ, 128, 0, stream>>>(qlc, query, 2, 0, emb, cat, BQ);
  gemm(cat, DM, gcnW16, DM, nullptr, 0, nullptr, 0, QN + 0, DM, gcnBias, gcnB,
       BQ, DDIM, 1);
  neighbor_kernel<<<BQ, 128, 0, stream>>>(qrc, query, 2, 1, emb, cat, BQ);
  gemm(cat, DM, gcnW16, DM, nullptr, 0, nullptr, 0, QN + DDIM, DM, gcnBias,
       gcnB, BQ, DDIM, 1);
  neighbor_kernel<<<FEWN, 128, 0, stream>>>(slc, support, 2, 0, emb, cat, FEWN);
  gemm(cat, DM, gcnW16, DM, nullptr, 0, nullptr, 0, SN + 0, DM, gcnBias, gcnB,
       FEWN, DDIM, 1);
  neighbor_kernel<<<FEWN, 128, 0, stream>>>(srcn, support, 2, 1, emb, cat,
                                            FEWN);
  gemm(cat, DM, gcnW16, DM, nullptr, 0, nullptr, 0, SN + DDIM, DM, gcnBias,
       gcnB, FEWN, DDIM, 1);

  // ---- support encoder (MLP + residual + LN) ----
  gemm(QN, DM, w116, DM, nullptr, 0, nullptr, 0, Hq, DF, b1, nullptr, BQ, DF,
       2);
  gemm(Hq, DF, w216, DF, nullptr, 0, nullptr, 0, SEh, DM, b2, nullptr, BQ, DM,
       0);
  ln_kernel<<<BQ, 32, 0, stream>>>(SEh, QN, lng, lnb, QG, BQ);
  gemm(SN, DM, w116, DM, nullptr, 0, nullptr, 0, Hs, DF, b1, nullptr, FEWN, DF,
       2);
  gemm(Hs, DF, w216, DF, nullptr, 0, nullptr, 0, SEhs, DM, b2, nullptr, FEWN,
       DM, 0);
  ln_kernel<<<FEWN, 32, 0, stream>>>(SEhs, SN, lng, lnb, SES, FEWN);
  support_mean_kernel<<<1, 256, 0, stream>>>(SES, SG, SGN);

  // ---- query encoder (2 LSTM steps; attention collapses to support_g) ----
  for (int step = 0; step < 2; ++step) {
    if (step == 0) {
      gemm(QG, DM, Wih16, DM, nullptr, 0, nullptr, 0, gates, 4 * HDIM, bih,
           bhh, BQ, 4 * HDIM, 0);
    } else {  // fused dual-pair GEMM: QG@Wih^T + hr@Whh^T in one pass
      gemm(QG, DM, Wih16, DM, hr, HDIM, Whh16, HDIM, gates, 4 * HDIM, bih, bhh,
           BQ, 4 * HDIM, 0);
    }
    long total = (long)BQ * HDIM;
    lstm_pw_kernel<<<(int)((total + 255) / 256), 256, 0, stream>>>(
        gates, cst, QG, SG, hbuf, hr, step, BQ);
  }

  final_kernel<<<BQ, 32, 0, stream>>>(hbuf, SGN, (float*)d_out, BQ);
}
